// NodeSAGELSTM_14302241096331
// MI455X (gfx1250) — matrix operand order repaired and checked by reading, not compiled, not run on hardware
//
#include <hip/hip_runtime.h>
#include <hip/hip_bf16.h>
#include <stdint.h>

// ---------------------------------------------------------------------------
// CDNA5 (gfx1250) implementation of SAGEConv x2 + LSTM(proj_size=3).
// GEMMs run on the WMMA pipe in bf16 (f32 accumulate); scatter-mean uses
// agent-scope fp32 global atomics; the serial LSTM scan runs on one wave32.
// ---------------------------------------------------------------------------

typedef __attribute__((ext_vector_type(16))) __bf16 v16bf;
typedef __attribute__((ext_vector_type(8)))  float  v8f;

#define BM   64      // M rows per block
#define LDT  136     // padded LDS stride (bf16 elems): 272B rows -> 16B-aligned
                     // fragment chunks, rows 4 banks apart (conflict-light)

// ---------------- small elementwise helpers ----------------

__global__ void cvt_bf16_kernel(const float* __restrict__ s, __bf16* __restrict__ d, int n) {
    int i = blockIdx.x * blockDim.x + threadIdx.x;
    if (i < n) d[i] = (__bf16)s[i];
}

__global__ void add_vec_kernel(const float* __restrict__ a, const float* __restrict__ b,
                               float* __restrict__ o, int n) {
    int i = blockIdx.x * blockDim.x + threadIdx.x;
    if (i < n) o[i] = a[i] + b[i];
}

__global__ void zero_f32_kernel(float* __restrict__ p, int n) {
    int i = blockIdx.x * blockDim.x + threadIdx.x;
    if (i < n) p[i] = 0.0f;
}

// ---------------- scatter-mean (gather + atomic segment sum) ----------------

__global__ void scatter_kernel(const float* __restrict__ xp,
                               const long long* __restrict__ src,
                               const long long* __restrict__ dst,
                               float* __restrict__ agg, float* __restrict__ cnt, int E) {
    int g = blockIdx.x * blockDim.x + threadIdx.x;
    int e = g >> 5;
    int lane = g & 31;
    if (e >= E) return;
    int s = (int)src[e];
    int d = (int)dst[e];
    float4 v = *(const float4*)(xp + (size_t)s * 128 + lane * 4);
    float* ap = agg + (size_t)d * 128 + lane * 4;
    __hip_atomic_fetch_add(ap + 0, v.x, __ATOMIC_RELAXED, __HIP_MEMORY_SCOPE_AGENT);
    __hip_atomic_fetch_add(ap + 1, v.y, __ATOMIC_RELAXED, __HIP_MEMORY_SCOPE_AGENT);
    __hip_atomic_fetch_add(ap + 2, v.z, __ATOMIC_RELAXED, __HIP_MEMORY_SCOPE_AGENT);
    __hip_atomic_fetch_add(ap + 3, v.w, __ATOMIC_RELAXED, __HIP_MEMORY_SCOPE_AGENT);
    if (lane == 0)
        __hip_atomic_fetch_add(cnt + d, 1.0f, __ATOMIC_RELAXED, __HIP_MEMORY_SCOPE_AGENT);
}

// ---------------- WMMA fragment loaders (contiguous k = 16*half + s, wave32) -------

__device__ __forceinline__ v16bf load_a_frag(const __bf16* As, int ms, int ks, int half, int lrow) {
    // A 16x32 bf16: lanes 0-15 M=0-15 hold K 0..15; lanes 16-31 hold K 16..31
    const __bf16* rp = As + (ms * 16 + lrow) * LDT + ks * 32;
    v16bf a;
#pragma unroll
    for (int r = 0; r < 4; ++r) {
        a[2 * r]         = rp[half * 16 + 2 * r];
        a[2 * r + 1]     = rp[half * 16 + 2 * r + 1];
        a[8 + 2 * r]     = rp[half * 16 + 8 + 2 * r];
        a[8 + 2 * r + 1] = rp[half * 16 + 8 + 2 * r + 1];
    }
    return a;
}

__device__ __forceinline__ v16bf load_b_frag(const __bf16* Bs, int nt, int ks, int half, int lrow) {
    // B 32x16 bf16: lane = output column; half-wave splits K 0..15 / 16..31.
    // Bs holds W row-major ([ncol][K]) == B^T, so Bs[col][k] = B[k][col].
    const __bf16* rp = Bs + (nt * 16 + lrow) * LDT + ks * 32 + half * 16;
    v16bf b;
#pragma unroll
    for (int r = 0; r < 16; ++r) b[r] = rp[r];
    return b;
}

// ---------------- fused WMMA GEMM ----------------
// OUT[m, ncolBase + n] = act( A1'(m,:) @ B1(n,:)^T  [+ A2(m,:) @ B2(n,:)^T]  + bias[n] )
// where A1' = A1 / max(cnt,1) when cnt != null (mean aggregation fused in).
// DUAL templated out so the inner loop is a straight WMMA accumulation chain.

template <bool DUAL>
__global__ __launch_bounds__(256) void wmma_gemm_kernel(
    const float* __restrict__ A1, const float* __restrict__ cnt,
    const float* __restrict__ A2,
    const __bf16* __restrict__ B1, const __bf16* __restrict__ B2,
    const float* __restrict__ bias, float* __restrict__ OUT,
    int M, int ldout, int relu) {
    extern __shared__ __bf16 smem[];
    __bf16* AS1 = smem;
    __bf16* AS2 = AS1 + BM * LDT;
    __bf16* BS1 = AS2 + BM * LDT;
    __bf16* BS2 = BS1 + 128 * LDT;

    const int tid = threadIdx.x;
    const int mbase = blockIdx.x * BM;
    const int ncolBase = blockIdx.y * 128;

    __builtin_prefetch(B1 + (size_t)ncolBase * 128, 0, 3);   // global_prefetch_b8
    if (DUAL) __builtin_prefetch(B2, 0, 3);

    // ---- stage A1 tile (f32 -> bf16, mean fused, OOB rows zero) ----
    for (int i = tid; i < BM * 32; i += 256) {
        int r = i >> 5, c4 = (i & 31) * 4;
        int row = mbase + r;
        float x0 = 0.f, x1 = 0.f, x2 = 0.f, x3 = 0.f;
        if (row < M) {
            const float* p = A1 + (size_t)row * 128 + c4;
            float s = 1.0f;
            if (cnt) s = 1.0f / fmaxf(cnt[row], 1.0f);
            x0 = p[0] * s; x1 = p[1] * s; x2 = p[2] * s; x3 = p[3] * s;
        }
        __bf16* d = AS1 + r * LDT + c4;   // 8B-aligned group of 4
        d[0] = (__bf16)x0; d[1] = (__bf16)x1; d[2] = (__bf16)x2; d[3] = (__bf16)x3;
    }
    if (DUAL) {
        for (int i = tid; i < BM * 32; i += 256) {
            int r = i >> 5, c4 = (i & 31) * 4;
            int row = mbase + r;
            float x0 = 0.f, x1 = 0.f, x2 = 0.f, x3 = 0.f;
            if (row < M) {
                const float* p = A2 + (size_t)row * 128 + c4;
                x0 = p[0]; x1 = p[1]; x2 = p[2]; x3 = p[3];
            }
            __bf16* d = AS2 + r * LDT + c4;
            d[0] = (__bf16)x0; d[1] = (__bf16)x1; d[2] = (__bf16)x2; d[3] = (__bf16)x3;
        }
    }
    // ---- stage B panels (already bf16), 16B vector copies ----
    for (int i = tid; i < 128 * 16; i += 256) {
        int r = i >> 4, c8 = (i & 15) * 8;
        *(uint4*)(BS1 + r * LDT + c8) =
            *(const uint4*)(B1 + (size_t)(ncolBase + r) * 128 + c8);
    }
    if (DUAL) {
        for (int i = tid; i < 128 * 16; i += 256) {
            int r = i >> 4, c8 = (i & 15) * 8;
            *(uint4*)(BS2 + r * LDT + c8) =
                *(const uint4*)(B2 + (size_t)(ncolBase + r) * 128 + c8);
        }
    }
    __syncthreads();

    const int wave = tid >> 5;
    const int lane = tid & 31;
    const int half = lane >> 4;
    const int lrow = lane & 15;
    const int nt = wave;   // each wave owns one 16-col tile of the 128-col panel

#pragma unroll
    for (int ms = 0; ms < BM / 16; ++ms) {
        v8f acc = {0.f, 0.f, 0.f, 0.f, 0.f, 0.f, 0.f, 0.f};
#pragma unroll
        for (int ks = 0; ks < 4; ++ks) {
            v16bf a = load_a_frag(AS1, ms, ks, half, lrow);
            v16bf b = load_b_frag(BS1, nt, ks, half, lrow);
            acc = __builtin_amdgcn_wmma_f32_16x16x32_bf16(
                false, a, false, b, (short)0, acc, false, false);
            if (DUAL) {
                v16bf a2 = load_a_frag(AS2, ms, ks, half, lrow);
                v16bf b2 = load_b_frag(BS2, nt, ks, half, lrow);
                acc = __builtin_amdgcn_wmma_f32_16x16x32_bf16(
                    false, a2, false, b2, (short)0, acc, false, false);
            }
        }
        // C/D layout: lane = column; VGPR v -> row half*8+v
        int n = ncolBase + nt * 16 + lrow;
        float bv = bias ? bias[n] : 0.0f;
        int mrow0 = mbase + ms * 16 + half * 8;
#pragma unroll
        for (int v = 0; v < 8; ++v) {
            int row = mrow0 + v;
            if (row < M) {
                float o = acc[v] + bv;
                if (relu) o = fmaxf(o, 0.0f);
                OUT[(size_t)row * ldout + n] = o;
            }
        }
    }
}

// ---------------- serial LSTM scan (one wave32, H=128, proj P=3) -----------

__device__ __forceinline__ float sigm(float x) { return 1.0f / (1.0f + __expf(-x)); }
__device__ __forceinline__ float tanh_fast(float x) {
    float e = __expf(2.0f * x);
    return (e - 1.0f) / (e + 1.0f);
}

__global__ void lstm_scan_kernel(const float* __restrict__ pre,
                                 const float* __restrict__ Whh,   // [512,3]
                                 const float* __restrict__ Whr,   // [3,128]
                                 float* __restrict__ out, int N) {
    const int lane = threadIdx.x;   // 32 lanes, each owns 4 cell lanes
    float whh_i[4][3], whh_f[4][3], whh_g[4][3], whh_o[4][3], whr[4][3];
    float c[4] = {0.f, 0.f, 0.f, 0.f};
#pragma unroll
    for (int q = 0; q < 4; ++q) {
        int hIdx = lane * 4 + q;
#pragma unroll
        for (int p = 0; p < 3; ++p) {
            whh_i[q][p] = Whh[(0   + hIdx) * 3 + p];
            whh_f[q][p] = Whh[(128 + hIdx) * 3 + p];
            whh_g[q][p] = Whh[(256 + hIdx) * 3 + p];
            whh_o[q][p] = Whh[(384 + hIdx) * 3 + p];
            whr[q][p]   = Whr[p * 128 + hIdx];
        }
    }
    float h0 = 0.f, h1 = 0.f, h2 = 0.f;
    for (int t = 0; t < N; ++t) {
        const float* pt = pre + (size_t)t * 512;
        __builtin_prefetch(pt + 512 + lane * 4, 0, 1);   // hide next step's latency
        float4 pi = *(const float4*)(pt +   0 + lane * 4);
        float4 pf = *(const float4*)(pt + 128 + lane * 4);
        float4 pg = *(const float4*)(pt + 256 + lane * 4);
        float4 po = *(const float4*)(pt + 384 + lane * 4);
        float ia[4] = {pi.x, pi.y, pi.z, pi.w};
        float fa[4] = {pf.x, pf.y, pf.z, pf.w};
        float ga[4] = {pg.x, pg.y, pg.z, pg.w};
        float oa[4] = {po.x, po.y, po.z, po.w};
        float a0 = 0.f, a1 = 0.f, a2 = 0.f;
#pragma unroll
        for (int q = 0; q < 4; ++q) {
            float gi = ia[q] + whh_i[q][0] * h0 + whh_i[q][1] * h1 + whh_i[q][2] * h2;
            float gf = fa[q] + whh_f[q][0] * h0 + whh_f[q][1] * h1 + whh_f[q][2] * h2;
            float gg = ga[q] + whh_g[q][0] * h0 + whh_g[q][1] * h1 + whh_g[q][2] * h2;
            float go = oa[q] + whh_o[q][0] * h0 + whh_o[q][1] * h1 + whh_o[q][2] * h2;
            c[q] = sigm(gf) * c[q] + sigm(gi) * tanh_fast(gg);
            float hr = sigm(go) * tanh_fast(c[q]);
            a0 += whr[q][0] * hr; a1 += whr[q][1] * hr; a2 += whr[q][2] * hr;
        }
#pragma unroll
        for (int off = 16; off > 0; off >>= 1) {        // wave32 butterfly reduce
            a0 += __shfl_xor(a0, off, 32);
            a1 += __shfl_xor(a1, off, 32);
            a2 += __shfl_xor(a2, off, 32);
        }
        h0 = a0; h1 = a1; h2 = a2;
        if (lane == 0) {
            out[(size_t)t * 3 + 0] = h0;
            out[(size_t)t * 3 + 1] = h1;
            out[(size_t)t * 3 + 2] = h2;
        }
    }
    if (lane == 0) {   // hN
        out[(size_t)N * 3 + 0] = h0;
        out[(size_t)N * 3 + 1] = h1;
        out[(size_t)N * 3 + 2] = h2;
    }
#pragma unroll
    for (int q = 0; q < 4; ++q)   // cN
        out[(size_t)N * 3 + 3 + lane * 4 + q] = c[q];
}

// ---------------- host orchestration ----------------

extern "C" void kernel_launch(void* const* d_in, const int* in_sizes, int n_in,
                              void* d_out, int out_size, void* d_ws, size_t ws_size,
                              hipStream_t stream) {
    const float*     x   = (const float*)d_in[0];
    const long long* ei  = (const long long*)d_in[1];
    const float* Wp1 = (const float*)d_in[2];
    const float* bp1 = (const float*)d_in[3];
    const float* Wl1 = (const float*)d_in[4];
    const float* bl1 = (const float*)d_in[5];
    const float* Wr1 = (const float*)d_in[6];
    const float* Wp2 = (const float*)d_in[7];
    const float* bp2 = (const float*)d_in[8];
    const float* Wl2 = (const float*)d_in[9];
    const float* bl2 = (const float*)d_in[10];
    const float* Wr2 = (const float*)d_in[11];
    const float* Wih = (const float*)d_in[12];
    const float* Whh = (const float*)d_in[13];
    const float* bih = (const float*)d_in[14];
    const float* bhh = (const float*)d_in[15];
    const float* Whr = (const float*)d_in[16];
    float* out = (float*)d_out;

    const int D = 128;
    const int N = in_sizes[0] / D;          // 50000
    const int E = in_sizes[1] / 2;          // 600000
    const long long* src = ei;
    const long long* dst = ei + E;

    char* ws = (char*)d_ws;
    size_t off = 0;
    auto alloc = [&](size_t bytes) -> void* {
        void* p = ws + off;
        off += (bytes + 255) & ~(size_t)255;
        return p;
    };
    float*  xp     = (float*)alloc((size_t)N * D * 4);       // projected feats (reused conv1/conv2)
    float*  agg    = (float*)alloc((size_t)N * D * 4);       // segment sum (reused)
    float*  cntv   = (float*)alloc((size_t)N * 4);           // segment count (reused)
    float*  h1     = (float*)alloc((size_t)N * D * 4);
    float*  h2     = (float*)alloc((size_t)N * D * 4);
    float*  pre    = (float*)alloc((size_t)N * 512 * 4);
    __bf16* Wp1b   = (__bf16*)alloc(16384 * 2);
    __bf16* Wl1b   = (__bf16*)alloc(16384 * 2);
    __bf16* Wr1b   = (__bf16*)alloc(16384 * 2);
    __bf16* Wp2b   = (__bf16*)alloc(16384 * 2);
    __bf16* Wl2b   = (__bf16*)alloc(16384 * 2);
    __bf16* Wr2b   = (__bf16*)alloc(16384 * 2);
    __bf16* Wihb   = (__bf16*)alloc(65536 * 2);
    float*  biasIH = (float*)alloc(512 * 4);

    // weight downconversion (one-time per call, tiny)
    cvt_bf16_kernel<<<64, 256, 0, stream>>>(Wp1, Wp1b, 16384);
    cvt_bf16_kernel<<<64, 256, 0, stream>>>(Wl1, Wl1b, 16384);
    cvt_bf16_kernel<<<64, 256, 0, stream>>>(Wr1, Wr1b, 16384);
    cvt_bf16_kernel<<<64, 256, 0, stream>>>(Wp2, Wp2b, 16384);
    cvt_bf16_kernel<<<64, 256, 0, stream>>>(Wl2, Wl2b, 16384);
    cvt_bf16_kernel<<<64, 256, 0, stream>>>(Wr2, Wr2b, 16384);
    cvt_bf16_kernel<<<256, 256, 0, stream>>>(Wih, Wihb, 65536);
    add_vec_kernel<<<2, 256, 0, stream>>>(bih, bhh, biasIH, 512);

    const size_t smem = (size_t)(2 * BM * LDT + 2 * 128 * LDT) * sizeof(__bf16); // ~102 KB
    dim3 gM((N + BM - 1) / BM, 1);
    dim3 gPre((N + BM - 1) / BM, 4);       // 512 output cols = 4 panels of 128
    int zgridA = (N * D + 255) / 256, zgridC = (N + 255) / 256;
    int sgrid = (E * 32 + 255) / 256;

    // ---- SAGEConv 1 ----
    zero_f32_kernel<<<zgridA, 256, 0, stream>>>(agg, N * D);
    zero_f32_kernel<<<zgridC, 256, 0, stream>>>(cntv, N);
    wmma_gemm_kernel<false><<<gM, 256, smem, stream>>>(x, nullptr, nullptr, Wp1b, nullptr,
                                                       bp1, xp, N, D, /*relu=*/1);
    scatter_kernel<<<sgrid, 256, 0, stream>>>(xp, src, dst, agg, cntv, E);
    wmma_gemm_kernel<true><<<gM, 256, smem, stream>>>(agg, cntv, x, Wl1b, Wr1b,
                                                      bl1, h1, N, D, /*relu=*/1);

    // ---- SAGEConv 2 ----
    zero_f32_kernel<<<zgridA, 256, 0, stream>>>(agg, N * D);
    zero_f32_kernel<<<zgridC, 256, 0, stream>>>(cntv, N);
    wmma_gemm_kernel<false><<<gM, 256, smem, stream>>>(h1, nullptr, nullptr, Wp2b, nullptr,
                                                       bp2, xp, N, D, /*relu=*/1);
    scatter_kernel<<<sgrid, 256, 0, stream>>>(xp, src, dst, agg, cntv, E);
    wmma_gemm_kernel<true><<<gM, 256, smem, stream>>>(agg, cntv, h1, Wl2b, Wr2b,
                                                      bl2, h2, N, D, /*relu=*/0);

    // ---- LSTM: input GEMM (parallel) + serial scan ----
    wmma_gemm_kernel<false><<<gPre, 256, smem, stream>>>(h2, nullptr, nullptr, Wihb, nullptr,
                                                         biasIH, pre, N, 512, /*relu=*/0);
    lstm_scan_kernel<<<1, 32, 0, stream>>>(pre, Whh, Whr, out, N);
}
